// GCN_single_42674795053976
// MI455X (gfx1250) — matrix operand order repaired and checked by reading, not compiled, not run on hardware
//
#include <hip/hip_runtime.h>

#define NN 131072      // total nodes
#define DD 128         // node dim
#define EE 1048576     // edges
#define GG 2048        // graphs (64 nodes each)

typedef __attribute__((ext_vector_type(16))) __bf16 v16bf;
typedef __attribute__((ext_vector_type(8)))  float  v8f;

// ---------------------------------------------------------------------------
// h[n,d] = x[n] * Wn[d] + bn[d]
// ---------------------------------------------------------------------------
__global__ void __launch_bounds__(256)
gcn_node_linear(const float* __restrict__ x, const float* __restrict__ Wn,
                const float* __restrict__ bn, float* __restrict__ h) {
    size_t idx = (size_t)blockIdx.x * 256 + threadIdx.x;
    int n = (int)(idx >> 7);
    int d = (int)(idx & 127);
    h[idx] = x[n] * Wn[d] + bn[d];
}

__global__ void __launch_bounds__(256)
gcn_zero(float* __restrict__ p, int n) {
    int i = blockIdx.x * 256 + threadIdx.x;
    if (i < n) p[i] = 0.0f;
}

// deg[t-1][col[e]] += 1  for each edge of type t
__global__ void __launch_bounds__(256)
gcn_degree(const int* __restrict__ col, const int* __restrict__ types,
           float* __restrict__ deg) {
    int e = blockIdx.x * 256 + threadIdx.x;
    if (e >= EE) return;
    int t = types[e];                       // 1..4
    unsafeAtomicAdd(&deg[(size_t)(t - 1) * NN + col[e]], 1.0f);
}

__global__ void __launch_bounds__(256)
gcn_dinv(const float* __restrict__ deg, float* __restrict__ dinv) {
    int i = blockIdx.x * 256 + threadIdx.x;
    if (i < 4 * NN) dinv[i] = rsqrtf(deg[i] + 1.0f);   // +1 self loop
}

// ---------------------------------------------------------------------------
// WMMA GEMM:  XW[N,128] = H[N,128] @ W[128,128]   (bf16 inputs, f32 accum)
// Block: 256 threads (8 waves). Each wave: 16 rows x 128 cols.
// W is repacked into LDS in exact B-fragment order once per block.
// ---------------------------------------------------------------------------
__global__ void __launch_bounds__(256)
gcn_gemm_wmma(const float* __restrict__ H, const float* __restrict__ W,
              float* __restrict__ XW) {
    // sB[kc][nt][lane][i] : B fragment for k-chunk kc, n-tile nt.
    // Element i of lane L holds W[kc*32 + (L>>4)*16 + i][nt*16 + (L&15)].
    __shared__ __align__(32) __bf16 sB[4 * 8 * 32 * 16];   // 32 KB

    const int tid = threadIdx.x;
    for (int idx = tid; idx < 4 * 8 * 32 * 16; idx += 256) {
        int i  = idx & 15;
        int L  = (idx >> 4) & 31;
        int nt = (idx >> 9) & 7;
        int kc = idx >> 12;
        int k  = kc * 32 + (L >> 4) * 16 + i;
        int n  = nt * 16 + (L & 15);
        sB[idx] = (__bf16)W[k * DD + n];
    }
    __syncthreads();

    const int wave = tid >> 5;
    const int lane = tid & 31;
    const int hlf  = lane >> 4;                 // lane half (0/1)
    const int mrow = blockIdx.x * 128 + wave * 16 + (lane & 15);
    const float* __restrict__ hrow = H + (size_t)mrow * DD;

    v8f acc[8];
#pragma unroll
    for (int nt = 0; nt < 8; ++nt) acc[nt] = (v8f){};

#pragma unroll
    for (int kc = 0; kc < 4; ++kc) {
        // A fragment (contiguous K order, matches sB): lane half h -> K {kb..kb+15}, kb = kc*32 + 16h
        const int kb = kc * 32 + hlf * 16;
        float4 f0 = *(const float4*)(hrow + kb);
        float4 f1 = *(const float4*)(hrow + kb + 4);
        float4 f2 = *(const float4*)(hrow + kb + 8);
        float4 f3 = *(const float4*)(hrow + kb + 12);
        v16bf a;
        a[0]  = (__bf16)f0.x; a[1]  = (__bf16)f0.y; a[2]  = (__bf16)f0.z; a[3]  = (__bf16)f0.w;
        a[4]  = (__bf16)f1.x; a[5]  = (__bf16)f1.y; a[6]  = (__bf16)f1.z; a[7]  = (__bf16)f1.w;
        a[8]  = (__bf16)f2.x; a[9]  = (__bf16)f2.y; a[10] = (__bf16)f2.z; a[11] = (__bf16)f2.w;
        a[12] = (__bf16)f3.x; a[13] = (__bf16)f3.y; a[14] = (__bf16)f3.z; a[15] = (__bf16)f3.w;

        const __bf16* bbase = sB + ((size_t)(kc * 8) * 32 + lane) * 16;
#pragma unroll
        for (int nt = 0; nt < 8; ++nt) {
            v16bf b = *(const v16bf*)(bbase + (size_t)nt * 32 * 16);
            acc[nt] = __builtin_amdgcn_wmma_f32_16x16x32_bf16(
                false, a, false, b, (short)0, acc[nt], false, false);
        }
    }

    // C/D 16x16 f32 layout: VGPR r -> M = r + 8*hlf, N = lane&15
    const int m0 = blockIdx.x * 128 + wave * 16 + hlf * 8;
#pragma unroll
    for (int nt = 0; nt < 8; ++nt) {
        int n = nt * 16 + (lane & 15);
#pragma unroll
        for (int r = 0; r < 8; ++r)
            XW[(size_t)(m0 + r) * DD + n] = acc[nt][r];
    }
}

// ---------------------------------------------------------------------------
// agg[n,:] = xw[n,:] * dinv_t[n]^2 + b     (self-loop message + bias)
// ---------------------------------------------------------------------------
__global__ void __launch_bounds__(256)
gcn_init_agg(const float* __restrict__ xw, const float* __restrict__ dinv_t,
             const float* __restrict__ b, float* __restrict__ agg) {
    size_t idx = (size_t)blockIdx.x * 256 + threadIdx.x;
    int n = (int)(idx >> 7);
    int d = (int)(idx & 127);
    float di = dinv_t[n];
    agg[idx] = xw[idx] * (di * di) + b[d];
}

// ---------------------------------------------------------------------------
// Edge scatter: one wave per edge; lane covers 4 contiguous dims (float4).
// agg[col,:] += xw[row,:] * dinv_t[row]*dinv_t[col]   (only edges of type t)
// ---------------------------------------------------------------------------
__global__ void __launch_bounds__(256)
gcn_scatter(const int* __restrict__ row, const int* __restrict__ col,
            const int* __restrict__ types, const float* __restrict__ dinv_t,
            const float* __restrict__ xw, float* __restrict__ agg, int t) {
    int gw   = (blockIdx.x * 256 + threadIdx.x) >> 5;   // global wave id == edge
    int lane = threadIdx.x & 31;
    if (gw >= EE) return;
    if (types[gw] != t) return;
    int r = row[gw];
    int c = col[gw];
    float norm = dinv_t[r] * dinv_t[c];
    const float4 v = *(const float4*)(xw + (size_t)r * DD + lane * 4);
    float* dst = agg + (size_t)c * DD + lane * 4;
    unsafeAtomicAdd(dst + 0, v.x * norm);
    unsafeAtomicAdd(dst + 1, v.y * norm);
    unsafeAtomicAdd(dst + 2, v.z * norm);
    unsafeAtomicAdd(dst + 3, v.w * norm);
}

// acc = (first ? 0 : acc) + relu(agg)
__global__ void __launch_bounds__(256)
gcn_accum(float* __restrict__ acc, const float* __restrict__ agg, int first) {
    size_t idx = (size_t)blockIdx.x * 256 + threadIdx.x;
    float v = fmaxf(agg[idx], 0.0f);
    acc[idx] = first ? v : (acc[idx] + v);
}

// pooled[g,d] = max over 64 nodes of h[g*64+i, d]
__global__ void __launch_bounds__(128)
gcn_pool(const float* __restrict__ h, float* __restrict__ pooled) {
    int g = blockIdx.x, d = threadIdx.x;
    const float* base = h + (size_t)g * 64 * DD + d;
    float m = base[0];
#pragma unroll 8
    for (int i = 1; i < 64; ++i) m = fmaxf(m, base[(size_t)i * DD]);
    pooled[(size_t)g * DD + d] = m;
}

// out[g] = (pooled[g,:] @ Wo1 + bo1) @ Wo2 + bo2
__global__ void __launch_bounds__(128)
gcn_mlp_out(const float* __restrict__ pooled, const float* __restrict__ Wo1,
            const float* __restrict__ bo1, const float* __restrict__ Wo2,
            const float* __restrict__ bo2, float* __restrict__ out) {
    __shared__ float sp[128];
    __shared__ float red[128];
    int g = blockIdx.x, j = threadIdx.x;
    sp[j] = pooled[(size_t)g * DD + j];
    __syncthreads();
    float t = bo1[j];
#pragma unroll 8
    for (int k = 0; k < 128; ++k) t += sp[k] * Wo1[k * DD + j];
    red[j] = t * Wo2[j];
    __syncthreads();
    for (int s = 64; s > 0; s >>= 1) {
        if (j < s) red[j] += red[j + s];
        __syncthreads();
    }
    if (j == 0) out[g] = red[0] + bo2[0];
}

// ---------------------------------------------------------------------------
extern "C" void kernel_launch(void* const* d_in, const int* in_sizes, int n_in,
                              void* d_out, int out_size, void* d_ws, size_t ws_size,
                              hipStream_t stream) {
    const float* x      = (const float*)d_in[0];
    const int*   eidx   = (const int*)d_in[1];
    const int*   row    = eidx;            // edge_index[0]
    const int*   col    = eidx + EE;       // edge_index[1]
    const int*   etypes = (const int*)d_in[2];
    const float* Wn     = (const float*)d_in[3];
    const float* bn     = (const float*)d_in[4];
    const float* Wt[4]  = {(const float*)d_in[5], (const float*)d_in[7],
                           (const float*)d_in[9], (const float*)d_in[11]};
    const float* bt[4]  = {(const float*)d_in[6], (const float*)d_in[8],
                           (const float*)d_in[10], (const float*)d_in[12]};
    const float* Wo1    = (const float*)d_in[13];
    const float* bo1    = (const float*)d_in[14];
    const float* Wo2    = (const float*)d_in[15];
    const float* bo2    = (const float*)d_in[16];
    float* out = (float*)d_out;

    // workspace layout (floats)
    const size_t ND = (size_t)NN * DD;     // 16,777,216
    float* ws   = (float*)d_ws;
    float* H    = ws;
    float* ACC  = H   + ND;
    float* XW   = ACC + ND;
    float* AGG  = XW  + ND;
    float* DEG  = AGG + ND;                // 4*NN
    float* DINV = DEG + (size_t)4 * NN;    // 4*NN
    float* POOL = DINV + (size_t)4 * NN;   // GG*DD

    const int nd_blocks  = (int)(ND / 256);        // 65536
    const int e_blocks   = EE / 256;               // 4096
    const int sc_blocks  = EE * 32 / 256;          // one wave per edge: 131072
    const int n4_blocks  = (4 * NN + 255) / 256;   // 2048

    // h = x @ Wn + bn
    gcn_node_linear<<<nd_blocks, 256, 0, stream>>>(x, Wn, bn, H);

    // per-type degrees and dinv (launch-invariant across layers)
    gcn_zero<<<n4_blocks, 256, 0, stream>>>(DEG, 4 * NN);
    gcn_degree<<<e_blocks, 256, 0, stream>>>(col, etypes, DEG);
    gcn_dinv<<<n4_blocks, 256, 0, stream>>>(DEG, DINV);

    float* hcur = H;
    float* hacc = ACC;
    for (int layer = 0; layer < 2; ++layer) {
        for (int t = 0; t < 4; ++t) {
            const float* dinv_t = DINV + (size_t)t * NN;
            gcn_gemm_wmma<<<NN / 128, 256, 0, stream>>>(hcur, Wt[t], XW);
            gcn_init_agg<<<nd_blocks, 256, 0, stream>>>(XW, dinv_t, bt[t], AGG);
            gcn_scatter<<<sc_blocks, 256, 0, stream>>>(row, col, etypes, dinv_t,
                                                       XW, AGG, t + 1);
            gcn_accum<<<nd_blocks, 256, 0, stream>>>(hacc, AGG, t == 0 ? 1 : 0);
        }
        // per-type terms are ReLU'd and >= 0, so relu(acc) == acc: just swap
        float* tmp = hcur; hcur = hacc; hacc = tmp;
    }

    gcn_pool<<<GG, 128, 0, stream>>>(hcur, POOL);
    gcn_mlp_out<<<GG, 128, 0, stream>>>(POOL, Wo1, bo1, Wo2, bo2, out);
}
